// Ours_total_77532749627656
// MI455X (gfx1250) — compile-verified, matrix operand order checked by reading, not run on hardware
//
#include <hip/hip_runtime.h>
#include <hip/hip_bf16.h>
#include <math.h>

typedef __bf16 bf16_t;
typedef __attribute__((ext_vector_type(16))) __bf16 v16bf;
typedef __attribute__((ext_vector_type(8)))  __bf16 v8bf;
typedef __attribute__((ext_vector_type(8)))  float  v8f;

#define EMBED    768
#define DEPTH    12
#define HEADS    12
#define HEAD_DIM 64
#define MLP_DIM  3072
#define NTOK     197
#define NPATCH   196
#define BATCH    32
#define TASKS    10
#define CLASSES  100
#define SPAD     208      // padded seq rows for Q buffer
#define KVPAD    224      // padded kv length (multiple of 32)
#define MROWS    (BATCH*NTOK)   // 6304
#define LEN_P    5

// ---------------- epilogue selectors ----------------
#define EPI_F32    0   // Cf[z*sC + m*ldc + n] = acc (+bias)
#define EPI_PATCH  1   // scatter patch-embed rows + pos_embed
#define EPI_RESID  2   // Cf[m*ldc+n] += acc + bias (in place residual)
#define EPI_GELU   3   // Cb[m*ldc+n] = bf16(gelu(acc+bias))
#define EPI_QKV    4   // scatter into Q / K / V^T head buffers
#define EPI_PKV_K  5   // prompt K rows -> K buffer slots 197..201
#define EPI_PKV_V  6   // prompt V rows -> V^T buffer slots 197..201
#define EPI_ATTN   7   // reassemble [b,s,h*64+d] as bf16

struct GemmParams {
  const bf16_t* A; const bf16_t* B; const float* bias;
  float* Cf; bf16_t* Cb;
  int M, N, K, lda, ldb, ldc;
  long sA, sB, sC;
  int epi;
  bf16_t *q, *kb, *vt;      // qkv scatter targets
  const float* pos;         // patch epilogue
};

// CDNA5 WMMA 16x32 bf16 fragment layout (A and B operands identical form):
// lane L: row = base_row + (L&15); halves: K[k0+h*8 .. +7] then K[k0+16+h*8 .. +7]
__device__ __forceinline__ v16bf frag_load_fast(const bf16_t* base, int row, int ld, int klo) {
  const v8bf x0 = *(const v8bf*)(base + (long)row * ld + klo);
  const v8bf x1 = *(const v8bf*)(base + (long)row * ld + klo + 16);
  v16bf f;
#pragma unroll
  for (int i = 0; i < 8; ++i) { f[i] = x0[i]; f[i + 8] = x1[i]; }
  return f;
}

__device__ __forceinline__ v16bf frag_load_guard(const bf16_t* base, int row, int rmax,
                                                 int ld, int klo) {
  v16bf f;
  if (row < rmax) {
    const v8bf x0 = *(const v8bf*)(base + (long)row * ld + klo);
    const v8bf x1 = *(const v8bf*)(base + (long)row * ld + klo + 16);
#pragma unroll
    for (int i = 0; i < 8; ++i) { f[i] = x0[i]; f[i + 8] = x1[i]; }
  } else {
#pragma unroll
    for (int i = 0; i < 16; ++i) f[i] = (__bf16)0.0f;
  }
  return f;
}

template<bool GUARD>
__device__ __forceinline__ void gemm_kloop(const GemmParams& p, const bf16_t* A, const bf16_t* B,
                                           int m0, int n0, int lr, int hi, v8f (&acc)[2][4]) {
  for (int k0 = 0; k0 < p.K; k0 += 32) {
    const int klo = k0 + hi * 8;
    v16bf a[2], b[4];
    if (GUARD) {
      a[0] = frag_load_guard(A, m0 + lr,      p.M, p.lda, klo);
      a[1] = frag_load_guard(A, m0 + 16 + lr, p.M, p.lda, klo);
#pragma unroll
      for (int nt = 0; nt < 4; ++nt)
        b[nt] = frag_load_guard(B, n0 + nt * 16 + lr, p.N, p.ldb, klo);
    } else {
      a[0] = frag_load_fast(A, m0 + lr,      p.lda, klo);
      a[1] = frag_load_fast(A, m0 + 16 + lr, p.lda, klo);
#pragma unroll
      for (int nt = 0; nt < 4; ++nt)
        b[nt] = frag_load_fast(B, n0 + nt * 16 + lr, p.ldb, klo);
      // speculative prefetch of next k-slab of the (weight) B operand
      __builtin_prefetch(B + (long)(n0 + lr) * p.ldb + klo + 32, 0, 1);
    }
#pragma unroll
    for (int mt = 0; mt < 2; ++mt)
#pragma unroll
      for (int nt = 0; nt < 4; ++nt)
        acc[mt][nt] = __builtin_amdgcn_wmma_f32_16x16x32_bf16(
            false, a[mt], false, b[nt], (short)0, acc[mt][nt], false, false);
  }
}

// Generic C = A * B^T GEMM. A:[M,K] row-major, B:[N,K] row-major (K contiguous both).
// Block = 64 threads = 2 waves; each wave owns a 32x64 tile of a 64x64 block tile.
__global__ __launch_bounds__(64) void gemm_bt_kernel(GemmParams p) {
  const int lane = threadIdx.x & 31;
  const int wave = threadIdx.x >> 5;
  const int z    = blockIdx.z;
  const bf16_t* A = p.A + (long)z * p.sA;
  const bf16_t* B = p.B + (long)z * p.sB;
  const int m0 = blockIdx.y * 64 + wave * 32;
  const int n0 = blockIdx.x * 64;
  const int lr = lane & 15;
  const int hi = lane >> 4;

  v8f acc[2][4];
#pragma unroll
  for (int i = 0; i < 2; ++i)
#pragma unroll
    for (int j = 0; j < 4; ++j)
#pragma unroll
      for (int e = 0; e < 8; ++e) acc[i][j][e] = 0.0f;

  const bool full = (m0 + 32 <= p.M) && (n0 + 64 <= p.N);
  if (full) gemm_kloop<false>(p, A, B, m0, n0, lr, hi, acc);
  else      gemm_kloop<true >(p, A, B, m0, n0, lr, hi, acc);

  // C/D layout: lane L, vgpr r -> (M = m0 + mt*16 + r + 8*(L>>4), N = n0 + nt*16 + (L&15))
#pragma unroll
  for (int mt = 0; mt < 2; ++mt)
#pragma unroll
    for (int nt = 0; nt < 4; ++nt)
#pragma unroll
      for (int rr = 0; rr < 8; ++rr) {
        const int mrow = m0 + mt * 16 + rr + hi * 8;
        const int ncol = n0 + nt * 16 + lr;
        if (mrow >= p.M || ncol >= p.N) continue;
        const float a = acc[mt][nt][rr];
        switch (p.epi) {
          case EPI_F32: {
            long idx = (long)z * p.sC + (long)mrow * p.ldc + ncol;
            p.Cf[idx] = p.bias ? (a + p.bias[ncol]) : a;
          } break;
          case EPI_PATCH: {
            int b = mrow / NPATCH, pp = mrow - b * NPATCH;
            long dst = ((long)b * NTOK + 1 + pp) * EMBED + ncol;
            p.Cf[dst] = a + p.bias[ncol] + p.pos[(long)(1 + pp) * EMBED + ncol];
          } break;
          case EPI_RESID: {
            long idx = (long)mrow * p.ldc + ncol;
            p.Cf[idx] = p.Cf[idx] + a + p.bias[ncol];
          } break;
          case EPI_GELU: {
            float v = a + p.bias[ncol];
            float g = 0.5f * v * (1.0f + erff(v * 0.70710678118654752f));
            p.Cb[(long)mrow * p.ldc + ncol] = (bf16_t)g;
          } break;
          case EPI_QKV: {
            float v = a + p.bias[ncol];
            int b = mrow / NTOK, s = mrow - b * NTOK;
            if (ncol < EMBED) {
              int h = ncol >> 6, d = ncol & 63;
              p.q[(((long)(b * HEADS + h)) * SPAD + s) * HEAD_DIM + d] = (bf16_t)v;
            } else if (ncol < 2 * EMBED) {
              int j = ncol - EMBED, h = j >> 6, d = j & 63;
              p.kb[(((long)(b * HEADS + h)) * KVPAD + s) * HEAD_DIM + d] = (bf16_t)v;
            } else {
              int j = ncol - 2 * EMBED, h = j >> 6, d = j & 63;
              p.vt[(((long)(b * HEADS + h)) * HEAD_DIM + d) * KVPAD + s] = (bf16_t)v;
            }
          } break;
          case EPI_PKV_K: {
            float v = a + p.bias[ncol];
            int b = mrow / LEN_P, t = mrow - b * LEN_P;
            int h = ncol >> 6, d = ncol & 63;
            p.kb[(((long)(b * HEADS + h)) * KVPAD + NTOK + t) * HEAD_DIM + d] = (bf16_t)v;
          } break;
          case EPI_PKV_V: {
            float v = a + p.bias[ncol];
            int b = mrow / LEN_P, t = mrow - b * LEN_P;
            int h = ncol >> 6, d = ncol & 63;
            p.vt[(((long)(b * HEADS + h)) * HEAD_DIM + d) * KVPAD + NTOK + t] = (bf16_t)v;
          } break;
          case EPI_ATTN: {
            int zb = z / HEADS, zh = z - zb * HEADS;
            p.Cb[((long)(zb * NTOK + mrow)) * EMBED + zh * HEAD_DIM + ncol] = (bf16_t)a;
          } break;
        }
      }
}

// ---------------- support kernels ----------------
__global__ __launch_bounds__(256) void cvt_kernel(const float* __restrict__ s,
                                                  bf16_t* __restrict__ d, long n) {
  long i = (long)blockIdx.x * blockDim.x + threadIdx.x;
  if (i < n) d[i] = (bf16_t)s[i];
}

__global__ __launch_bounds__(256) void patchify_kernel(const float* __restrict__ in,
                                                       bf16_t* __restrict__ out) {
  long i = (long)blockIdx.x * blockDim.x + threadIdx.x;
  if (i >= (long)BATCH * NPATCH * EMBED) return;
  int c = (int)(i % EMBED);
  long r = i / EMBED;
  int p = (int)(r % NPATCH);
  int b = (int)(r / NPATCH);
  int ch = c >> 8, ii = (c >> 4) & 15, jj = c & 15;
  int py = p / 14, px = p - py * 14;
  out[i] = (bf16_t)in[(((long)(b * 3 + ch) * 224) + py * 16 + ii) * 224 + px * 16 + jj];
}

__global__ __launch_bounds__(256) void cls_pos_kernel(const float* __restrict__ cls,
                                                      const float* __restrict__ pos,
                                                      float* __restrict__ x0) {
  int i = blockIdx.x * blockDim.x + threadIdx.x;
  if (i >= BATCH * EMBED) return;
  int b = i / EMBED, c = i - b * EMBED;
  x0[(long)b * NTOK * EMBED + c] = cls[c] + pos[c];
}

__global__ __launch_bounds__(256) void ln_kernel(const float* __restrict__ x,
                                                 const float* __restrict__ w,
                                                 const float* __restrict__ b,
                                                 bf16_t* __restrict__ obf,
                                                 float* __restrict__ of) {
  const int row = blockIdx.x;
  const int t = threadIdx.x;
  const float* xr = x + (long)row * EMBED;
  float v0 = xr[t], v1 = xr[t + 256], v2 = xr[t + 512];
  __shared__ float red[256];
  red[t] = v0 + v1 + v2;
  __syncthreads();
  for (int o = 128; o > 0; o >>= 1) { if (t < o) red[t] += red[t + o]; __syncthreads(); }
  float mean = red[0] * (1.0f / EMBED);
  __syncthreads();
  float d0 = v0 - mean, d1 = v1 - mean, d2 = v2 - mean;
  red[t] = d0 * d0 + d1 * d1 + d2 * d2;
  __syncthreads();
  for (int o = 128; o > 0; o >>= 1) { if (t < o) red[t] += red[t + o]; __syncthreads(); }
  float rstd = rsqrtf(red[0] * (1.0f / EMBED) + 1e-6f);
  float o0 = d0 * rstd * w[t]       + b[t];
  float o1 = d1 * rstd * w[t + 256] + b[t + 256];
  float o2 = d2 * rstd * w[t + 512] + b[t + 512];
  long base = (long)row * EMBED;
  obf[base + t] = (bf16_t)o0; obf[base + t + 256] = (bf16_t)o1; obf[base + t + 512] = (bf16_t)o2;
  if (of) { of[base + t] = o0; of[base + t + 256] = o1; of[base + t + 512] = o2; }
}

__global__ __launch_bounds__(256) void softmax_kernel(const float* __restrict__ Pf,
                                                      bf16_t* __restrict__ Pb,
                                                      int kvlen, float scale) {
  const int s = blockIdx.x;     // 0..196
  const int z = blockIdx.y;     // 0..B*H-1
  const int t = threadIdx.x;
  long base = ((long)z * NTOK + s) * KVPAD;
  float v = (t < kvlen) ? Pf[base + t] * scale : -3.0e38f;
  __shared__ float red[256];
  red[t] = v;
  __syncthreads();
  for (int o = 128; o > 0; o >>= 1) { if (t < o) red[t] = fmaxf(red[t], red[t + o]); __syncthreads(); }
  float mx = red[0];
  __syncthreads();
  float e = (t < kvlen) ? __expf(v - mx) : 0.0f;
  red[t] = e;
  __syncthreads();
  for (int o = 128; o > 0; o >>= 1) { if (t < o) red[t] += red[t + o]; __syncthreads(); }
  float inv = 1.0f / red[0];
  if (t < KVPAD) Pb[base + t] = (bf16_t)(e * inv);   // zero-pads kv>=kvlen
}

__global__ __launch_bounds__(256) void topk_kernel(const float* __restrict__ x,
                                                   const float* __restrict__ keyp,
                                                   int* __restrict__ topk) {
  const int b = blockIdx.x, t = threadIdx.x;
  const float* q = x + (long)b * NTOK * EMBED;   // cls row
  __shared__ float red[256];
  float best = -3.0e38f; int bi = 0;
  for (int k = 0; k < TASKS; ++k) {
    const float* kr = keyp + (long)k * EMBED;
    float d = 0.0f, n = 0.0f;
    for (int i = t; i < EMBED; i += 256) { float kv = kr[i]; d += q[i] * kv; n += kv * kv; }
    red[t] = d; __syncthreads();
    for (int o = 128; o > 0; o >>= 1) { if (t < o) red[t] += red[t + o]; __syncthreads(); }
    float dot = red[0]; __syncthreads();
    red[t] = n; __syncthreads();
    for (int o = 128; o > 0; o >>= 1) { if (t < o) red[t] += red[t + o]; __syncthreads(); }
    float nrm = red[0]; __syncthreads();
    float score = dot * rsqrtf(nrm + 1e-12f);  // argmax cos == argmin (1-cos); |q| factor constant
    if (score > best) { best = score; bi = k; }
  }
  if (t == 0) topk[b] = bi;
}

__global__ __launch_bounds__(256) void build_prompts_kernel(const float* __restrict__ g_pr,
                                                            const float* __restrict__ e_pr,
                                                            const int* __restrict__ topk,
                                                            bf16_t* __restrict__ Pk,
                                                            bf16_t* __restrict__ Pv) {
  long i = (long)blockIdx.x * blockDim.x + threadIdx.x;
  const long total = (long)5 * 2 * BATCH * LEN_P * EMBED;
  if (i >= total) return;
  int c = (int)(i % EMBED); long r = i / EMBED;
  int t = (int)(r % LEN_P); r /= LEN_P;
  int b = (int)(r % BATCH); r /= BATCH;
  int kv = (int)(r % 2);   r /= 2;
  int layer = (int)r;                       // 0..4
  float val;
  if (layer < 2) {                          // g-prompts, layers 0/1
    int gi = (2 * layer + kv) * LEN_P + t;
    val = g_pr[(long)gi * EMBED + c];
  } else {                                  // e-prompts, layers 2/3/4
    int li = layer - 2;
    int ei = (2 * li + kv) * LEN_P + t;
    val = e_pr[((long)topk[b] * 30 + ei) * EMBED + c];
  }
  long dst = (((long)layer * BATCH + b) * LEN_P + t) * EMBED + c;
  (kv ? Pv : Pk)[dst] = (bf16_t)val;
}

__global__ __launch_bounds__(256) void final_out_kernel(const float* __restrict__ logits,
                                                        const float* __restrict__ maskp,
                                                        const int* __restrict__ topk,
                                                        float* __restrict__ out) {
  int i = blockIdx.x * blockDim.x + threadIdx.x;
  if (i >= BATCH * CLASSES) return;
  int b = i / CLASSES, c = i - b * CLASSES;
  float m = maskp[(long)topk[b] * CLASSES + c];
  out[i] = logits[i] * (1.0f / (1.0f + __expf(-m)));
}

// ---------------- host side ----------------
extern "C" void kernel_launch(void* const* d_in, const int* in_sizes, int n_in,
                              void* d_out, int out_size, void* d_ws, size_t ws_size,
                              hipStream_t stream) {
  const float* inputs  = (const float*)d_in[0];
  const float* patch_w = (const float*)d_in[1];
  const float* patch_b = (const float*)d_in[2];
  const float* cls_tok = (const float*)d_in[3];
  const float* pos_emb = (const float*)d_in[4];
  const float* ln1_w   = (const float*)d_in[5];
  const float* ln1_b   = (const float*)d_in[6];
  const float* qkv_w   = (const float*)d_in[7];
  const float* qkv_b   = (const float*)d_in[8];
  const float* proj_w  = (const float*)d_in[9];
  const float* proj_b  = (const float*)d_in[10];
  const float* ln2_w   = (const float*)d_in[11];
  const float* ln2_b   = (const float*)d_in[12];
  const float* fc1_w   = (const float*)d_in[13];
  const float* fc1_b   = (const float*)d_in[14];
  const float* fc2_w   = (const float*)d_in[15];
  const float* fc2_b   = (const float*)d_in[16];
  const float* norm_w  = (const float*)d_in[17];
  const float* norm_b  = (const float*)d_in[18];
  const float* head_w  = (const float*)d_in[19];
  const float* head_b  = (const float*)d_in[20];
  const float* keyp    = (const float*)d_in[21];
  const float* maskp   = (const float*)d_in[22];
  const float* g_pr    = (const float*)d_in[23];
  const float* e_pr    = (const float*)d_in[24];
  (void)in_sizes; (void)n_in; (void)out_size; (void)ws_size;

  float* out_logits_dst = (float*)d_out;                 // [32,100]
  float* feat_dst = (float*)d_out + BATCH * CLASSES;     // [32,197,768]

  char* wp = (char*)d_ws;
  auto alloc = [&](size_t bytes) -> char* {
    char* p = wp; wp += (bytes + 255) & ~(size_t)255; return p;
  };

  bf16_t* patchw_bf = (bf16_t*)alloc((size_t)EMBED * EMBED * 2);
  bf16_t* qkvw_bf   = (bf16_t*)alloc((size_t)DEPTH * 3 * EMBED * EMBED * 2);
  bf16_t* projw_bf  = (bf16_t*)alloc((size_t)DEPTH * EMBED * EMBED * 2);
  bf16_t* fc1w_bf   = (bf16_t*)alloc((size_t)DEPTH * MLP_DIM * EMBED * 2);
  bf16_t* fc2w_bf   = (bf16_t*)alloc((size_t)DEPTH * EMBED * MLP_DIM * 2);
  bf16_t* headw_bf  = (bf16_t*)alloc((size_t)CLASSES * EMBED * 2);
  bf16_t* patches   = (bf16_t*)alloc((size_t)BATCH * NPATCH * EMBED * 2);
  float*  x0        = (float*)alloc((size_t)MROWS * EMBED * 4);
  float*  x         = (float*)alloc((size_t)MROWS * EMBED * 4);
  bf16_t* xln_bf    = (bf16_t*)alloc((size_t)MROWS * EMBED * 2);   // LN out / attn-out reuse
  bf16_t* h_bf      = (bf16_t*)alloc((size_t)MROWS * MLP_DIM * 2);
  bf16_t* Qb        = (bf16_t*)alloc((size_t)BATCH * HEADS * SPAD * HEAD_DIM * 2);
  bf16_t* Kb        = (bf16_t*)alloc((size_t)BATCH * HEADS * KVPAD * HEAD_DIM * 2);
  bf16_t* Vt        = (bf16_t*)alloc((size_t)BATCH * HEADS * HEAD_DIM * KVPAD * 2);
  float*  Pf        = (float*)alloc((size_t)BATCH * HEADS * NTOK * KVPAD * 4);
  bf16_t* Pb        = (bf16_t*)alloc((size_t)BATCH * HEADS * NTOK * KVPAD * 2);
  bf16_t* Pk        = (bf16_t*)alloc((size_t)5 * BATCH * LEN_P * EMBED * 2);
  bf16_t* Pv        = (bf16_t*)alloc((size_t)5 * BATCH * LEN_P * EMBED * 2);
  float*  logits    = (float*)alloc((size_t)BATCH * CLASSES * 4);
  int*    topk      = (int*)alloc((size_t)BATCH * 4);

  auto gemm = [&](int epi, const bf16_t* A, const bf16_t* B, const float* bias,
                  float* Cf, bf16_t* Cb, int M, int N, int K,
                  int lda, int ldb, int ldc, long sA, long sB, long sC, int batch) {
    GemmParams p;
    p.A = A; p.B = B; p.bias = bias; p.Cf = Cf; p.Cb = Cb;
    p.M = M; p.N = N; p.K = K; p.lda = lda; p.ldb = ldb; p.ldc = ldc;
    p.sA = sA; p.sB = sB; p.sC = sC; p.epi = epi;
    p.q = Qb; p.kb = Kb; p.vt = Vt; p.pos = pos_emb;
    dim3 grid((N + 63) / 64, (M + 63) / 64, batch);
    gemm_bt_kernel<<<grid, 64, 0, stream>>>(p);
  };
  auto cvt = [&](const float* s, bf16_t* d, long n) {
    cvt_kernel<<<(int)((n + 255) / 256), 256, 0, stream>>>(s, d, n);
  };

  // ---- 0. weights fp32 -> bf16 (model fits in L2 as bf16) ----
  cvt(patch_w, patchw_bf, (long)EMBED * EMBED);
  cvt(qkv_w,   qkvw_bf,   (long)DEPTH * 3 * EMBED * EMBED);
  cvt(proj_w,  projw_bf,  (long)DEPTH * EMBED * EMBED);
  cvt(fc1_w,   fc1w_bf,   (long)DEPTH * MLP_DIM * EMBED);
  cvt(fc2_w,   fc2w_bf,   (long)DEPTH * EMBED * MLP_DIM);
  cvt(head_w,  headw_bf,  (long)CLASSES * EMBED);

  // ---- 1. patch embed + cls + pos ----
  {
    long n = (long)BATCH * NPATCH * EMBED;
    patchify_kernel<<<(int)((n + 255) / 256), 256, 0, stream>>>(inputs, patches);
    cls_pos_kernel<<<(BATCH * EMBED + 255) / 256, 256, 0, stream>>>(cls_tok, pos_emb, x0);
    gemm(EPI_PATCH, patches, patchw_bf, patch_b, x0, nullptr,
         BATCH * NPATCH, EMBED, EMBED, EMBED, EMBED, EMBED, 0, 0, 0, 1);
  }

  auto run_pass = [&](bool prompts) {
    for (int n = 0; n < DEPTH; ++n) {
      const int kvlen = (prompts && n < 5) ? (NTOK + LEN_P) : NTOK;
      const bf16_t* Wqkv = qkvw_bf + (long)n * 3 * EMBED * EMBED;
      // LN1 -> bf16
      ln_kernel<<<MROWS, 256, 0, stream>>>(x, ln1_w + (long)n * EMBED,
                                           ln1_b + (long)n * EMBED, xln_bf, nullptr);
      // QKV projection -> Q / K / V^T head buffers
      gemm(EPI_QKV, xln_bf, Wqkv, qkv_b + (long)n * 3 * EMBED, nullptr, nullptr,
           MROWS, 3 * EMBED, EMBED, EMBED, EMBED, 0, 0, 0, 0, 1);
      if (prompts && n < 5) {   // project prompt tokens through Wk / Wv
        const bf16_t* Apk = Pk + (long)n * BATCH * LEN_P * EMBED;
        const bf16_t* Apv = Pv + (long)n * BATCH * LEN_P * EMBED;
        gemm(EPI_PKV_K, Apk, Wqkv + (long)EMBED * EMBED,
             qkv_b + (long)n * 3 * EMBED + EMBED, nullptr, nullptr,
             BATCH * LEN_P, EMBED, EMBED, EMBED, EMBED, 0, 0, 0, 0, 1);
        gemm(EPI_PKV_V, Apv, Wqkv + (long)2 * EMBED * EMBED,
             qkv_b + (long)n * 3 * EMBED + 2 * EMBED, nullptr, nullptr,
             BATCH * LEN_P, EMBED, EMBED, EMBED, EMBED, 0, 0, 0, 0, 1);
      }
      // scores = Q * K^T  (batched over b*h)
      gemm(EPI_F32, Qb, Kb, nullptr, Pf, nullptr,
           NTOK, kvlen, HEAD_DIM, HEAD_DIM, HEAD_DIM, KVPAD,
           (long)SPAD * HEAD_DIM, (long)KVPAD * HEAD_DIM, (long)NTOK * KVPAD,
           BATCH * HEADS);
      // softmax (scaled), zero-padded bf16 probabilities
      softmax_kernel<<<dim3(NTOK, BATCH * HEADS), 256, 0, stream>>>(Pf, Pb, kvlen, 0.125f);
      // attn out = P * V  (V stored transposed -> same A*B^T form), write bf16 [b,s,h*d]
      gemm(EPI_ATTN, Pb, Vt, nullptr, nullptr, xln_bf,
           NTOK, HEAD_DIM, KVPAD, KVPAD, KVPAD, 0,
           (long)NTOK * KVPAD, (long)HEAD_DIM * KVPAD, 0, BATCH * HEADS);
      // proj + residual
      gemm(EPI_RESID, xln_bf, projw_bf + (long)n * EMBED * EMBED,
           proj_b + (long)n * EMBED, x, nullptr,
           MROWS, EMBED, EMBED, EMBED, EMBED, EMBED, 0, 0, 0, 1);
      // MLP
      ln_kernel<<<MROWS, 256, 0, stream>>>(x, ln2_w + (long)n * EMBED,
                                           ln2_b + (long)n * EMBED, xln_bf, nullptr);
      gemm(EPI_GELU, xln_bf, fc1w_bf + (long)n * MLP_DIM * EMBED,
           fc1_b + (long)n * MLP_DIM, nullptr, h_bf,
           MROWS, MLP_DIM, EMBED, EMBED, EMBED, MLP_DIM, 0, 0, 0, 1);
      gemm(EPI_RESID, h_bf, fc2w_bf + (long)n * EMBED * MLP_DIM,
           fc2_b + (long)n * EMBED, x, nullptr,
           MROWS, EMBED, MLP_DIM, MLP_DIM, MLP_DIM, EMBED, 0, 0, 0, 1);
    }
  };

  // ---- 2. query pass (no prompts) ----
  hipMemcpyAsync(x, x0, (size_t)MROWS * EMBED * 4, hipMemcpyDeviceToDevice, stream);
  run_pass(false);

  // ---- 3. prompt selection + gather ----
  topk_kernel<<<BATCH, 256, 0, stream>>>(x, keyp, topk);
  {
    long n = (long)5 * 2 * BATCH * LEN_P * EMBED;
    build_prompts_kernel<<<(int)((n + 255) / 256), 256, 0, stream>>>(g_pr, e_pr, topk, Pk, Pv);
  }

  // ---- 4. prompted pass ----
  hipMemcpyAsync(x, x0, (size_t)MROWS * EMBED * 4, hipMemcpyDeviceToDevice, stream);
  run_pass(true);

  // ---- 5. final LN (feat to d_out), head, masked output ----
  ln_kernel<<<MROWS, 256, 0, stream>>>(x, norm_w, norm_b, xln_bf, feat_dst);
  // head over cls rows: A row b = xln_bf row b*197 via lda = 197*768
  gemm(EPI_F32, xln_bf, headw_bf, head_b, logits, nullptr,
       BATCH, CLASSES, EMBED, NTOK * EMBED, EMBED, CLASSES, 0, 0, 0, 1);
  final_out_kernel<<<(BATCH * CLASSES + 255) / 256, 256, 0, stream>>>(logits, maskp, topk,
                                                                      out_logits_dst);
}
